// LlamaAttention_27848567947510
// MI455X (gfx1250) — matrix operand order repaired and checked by reading, not compiled, not run on hardware
//
#include <hip/hip_runtime.h>
#include <hip/hip_bf16.h>

// ---------------------------------------------------------------------------
// LLaMA attention layer for MI455X (gfx1250, wave32, WMMA + TDM).
// Pipeline: cvt(x,w*) -> GEMM(q,k,v) -> RoPE -> flash-attention -> GEMM(out)
// GEMM math in bf16 WMMA (V_WMMA_F32_16X16X32_BF16), fp32 accumulate.
// Row-major LDS tiles staged by the Tensor Data Mover (TENSOR_LOAD_TO_LDS,
// s_wait_tensorcnt); transposed tiles staged manually.
// ---------------------------------------------------------------------------

typedef __bf16 bf16;
typedef __attribute__((ext_vector_type(16))) __bf16 v16bf;
typedef __attribute__((ext_vector_type(4)))  __bf16 v4bf;
typedef __attribute__((ext_vector_type(8)))  float  v8f;
typedef __attribute__((ext_vector_type(4)))  float  v4f;
typedef __attribute__((ext_vector_type(4)))  unsigned int u32x4;
typedef __attribute__((ext_vector_type(8)))  int    i32x8;
typedef __attribute__((ext_vector_type(4)))  int    i32x4;

#define B_  2
#define S_  2048
#define D_  2048
#define H_  16
#define HD_ 128

#if defined(__has_builtin)
#  if __has_builtin(__builtin_amdgcn_tensor_load_to_lds) && \
      __has_builtin(__builtin_amdgcn_s_wait_tensorcnt)
#    define HAVE_TDM 1
#  endif
#endif
#ifndef HAVE_TDM
#  define HAVE_TDM 0
#endif

// ---------------------------------------------------------------------------
// TDM: 2-D tile load Global -> LDS. Descriptor per CDNA5 ISA
// §8.3 (group0) / §8.4 (group1). data_size=2 bytes (bf16). tile_w elems
// contiguous along dim0, tile_h lines of stride `stride_elems`.
// Issued by one wave; that wave must s_wait_tensorcnt(0) before the barrier.
// This toolchain's builtin is the 6-arg form:
//   (u32x4 g0, i32x8 g1, i32x4 g2, i32x4 g3, i32x8 g4, i32 cpol)
// ---------------------------------------------------------------------------
__device__ __forceinline__ unsigned lds_addr_of(const void* p) {
  // generic LDS pointer = {SHARED_BASE aperture, 32'lds_offset}
  return (unsigned)(unsigned long long)(uintptr_t)p;
}

#if HAVE_TDM
__device__ __forceinline__ void tdm_load_2d(unsigned lds_byte_addr, const void* gptr,
                                            unsigned tile_w, unsigned tile_h,
                                            unsigned long long stride_elems,
                                            unsigned tensor_w, unsigned tensor_h) {
  unsigned long long ga = (unsigned long long)(uintptr_t)gptr;
  u32x4 g0;
  g0[0] = 1u;                                   // count=1, user descriptor
  g0[1] = lds_byte_addr;                        // lds_addr [63:32]
  g0[2] = (unsigned)(ga & 0xffffffffull);       // global_addr[31:0]
  g0[3] = (unsigned)((ga >> 32) & 0x01ffffffull)// global_addr[56:32]
        | (2u << 30);                           // type=2 ("image")
  i32x8 g1;
  g1[0] = 1 << 16;                              // data_size=1 -> 2-byte elems
  g1[1] = (int)((tensor_w & 0xffffu) << 16);                      // tdim0 lo16
  g1[2] = (int)((tensor_w >> 16) | ((tensor_h & 0xffffu) << 16)); // tdim0 hi / tdim1 lo
  g1[3] = (int)((tensor_h >> 16) | ((tile_w & 0xffffu) << 16));   // tdim1 hi / tile_dim0
  g1[4] = (int)(tile_h & 0xffffu);              // tile_dim1 (tile_dim2 = 0 -> 2D)
  g1[5] = (int)(stride_elems & 0xffffffffull);  // tensor_dim0_stride lo32
  g1[6] = (int)((stride_elems >> 32) & 0xffffull); // stride hi16; dim1_stride lo = 0
  g1[7] = 0;
  i32x4 z4 = {0, 0, 0, 0};                      // groups 2/3 unused (<=2D)
  i32x8 z8 = {0, 0, 0, 0, 0, 0, 0, 0};
  __builtin_amdgcn_tensor_load_to_lds(g0, g1, z4, z4, z8, 0);
}
#endif

// ---------------------------------------------------------------------------
// Fragment loaders following CDNA5 ISA 16-bit WMMA VGPR layouts (CDNA5 ISA).
// A (16x32, MxK): lanes 0-15 rows, K {0..7,16..23}; lanes 16-31, K {8..15,24..31}.
// B (32x16, KxN): lane holds column N=lane&15; lane-half selects K {0..7,16..23}/{8..15,24..31}.
// B tiles stored N-major in LDS so each lane's K-values are two contiguous runs of 8.
// ---------------------------------------------------------------------------
__device__ __forceinline__ v16bf load_frag_a(const bf16* base, int row, int ld, int lane) {
  int r  = row + (lane & 15);
  int kh = (lane >> 4) & 1;
  const bf16* p = base + (size_t)r * ld + kh * 8;
  v16bf f;
  ((u32x4*)&f)[0] = *(const u32x4*)(p);
  ((u32x4*)&f)[1] = *(const u32x4*)(p + 16);
  return f;
}

__device__ __forceinline__ v16bf load_frag_b(const bf16* base, int ncol, int ld, int lane) {
  int n  = ncol + (lane & 15);
  int kh = (lane >> 4) & 1;
  const bf16* p = base + (size_t)n * ld + kh * 8;
  v16bf f;
  ((u32x4*)&f)[0] = *(const u32x4*)(p);
  ((u32x4*)&f)[1] = *(const u32x4*)(p + 16);
  return f;
}

__device__ __forceinline__ v8f wmma_bf16(v16bf a, v16bf b, v8f c) {
  return __builtin_amdgcn_wmma_f32_16x16x32_bf16(false, a, false, b, (short)0, c, false, false);
}

// ---------------------------------------------------------------------------
// fp32 -> bf16 conversion
// ---------------------------------------------------------------------------
__global__ void __launch_bounds__(256) cvt_f32_bf16(const float* __restrict__ src,
                                                    bf16* __restrict__ dst, int n4) {
  int i = blockIdx.x * 256 + threadIdx.x;
  if (i < n4) {
    v4f f = ((const v4f*)src)[i];
    v4bf o;
    o[0] = (bf16)f[0]; o[1] = (bf16)f[1]; o[2] = (bf16)f[2]; o[3] = (bf16)f[3];
    ((v4bf*)dst)[i] = o;
  }
}

// ---------------------------------------------------------------------------
// bf16 WMMA GEMM: C(MxN) = A(MxK) * W(KxN). Block tile 128x128, BK=64,
// 256 threads = 8 waves (2x4), wave tile 64x32, 16 WMMAs per wave per stage.
// A tile (row-major) staged by TDM; B tile transposed manually into LDS.
// ---------------------------------------------------------------------------
template <typename OutT>
__global__ void __launch_bounds__(256) gemm_bf16_wmma(const bf16* __restrict__ A,
                                                      const bf16* __restrict__ W,
                                                      OutT* __restrict__ C,
                                                      int M, int N, int K) {
  __shared__ bf16 As[128 * 64];   // [row][k]   row-major, 16KB
  __shared__ bf16 Bs[128 * 64];   // [n][k]     transposed (N-major), 16KB

  const int tid  = threadIdx.x;
  const int lane = tid & 31;
  const int wave = tid >> 5;
  const int bm = blockIdx.x * 128;
  const int bn = blockIdx.y * 128;
  const int wm = (wave & 1) * 64;
  const int wn = (wave >> 1) * 32;
#if HAVE_TDM
  const unsigned ldsA = lds_addr_of(&As[0]);
#endif

  v8f acc[4][2] = {};

  for (int k0 = 0; k0 < K; k0 += 64) {
    // ---- Stage A tile 128x64 (row-major) ----
#if HAVE_TDM
    if (wave == 0)
      tdm_load_2d(ldsA, &A[(size_t)bm * K + k0], /*tile_w=*/64, /*tile_h=*/128,
                  (unsigned long long)K, (unsigned)K, (unsigned)M);
#else
    #pragma unroll
    for (int it = 0; it < 4; ++it) {
      int idx = (it * 256 + tid) * 8;          // elem id in 128*64
      int r = idx >> 6, c = idx & 63;
      *(u32x4*)&As[idx] = *(const u32x4*)&A[(size_t)(bm + r) * K + k0 + c];
    }
#endif
    // ---- Stage B tile 64x128, transposed into Bs[n][k] ----
    #pragma unroll
    for (int it = 0; it < 4; ++it) {
      int idx = (it * 256 + tid) * 8;          // elem id in 64*128
      int kk = idx >> 7, n = idx & 127;
      u32x4 d = *(const u32x4*)&W[(size_t)(k0 + kk) * N + bn + n];
      const bf16* dp = (const bf16*)&d;
      #pragma unroll
      for (int e = 0; e < 8; ++e) Bs[(n + e) * 64 + kk] = dp[e];
    }
    if (k0 + 64 < K) {
      __builtin_prefetch(&A[(size_t)(bm + (tid >> 1)) * K + k0 + 64], 0, 0);
      __builtin_prefetch(&W[(size_t)(k0 + 64 + (tid >> 2)) * N + bn], 0, 0);
    }
#if HAVE_TDM
    if (wave == 0) __builtin_amdgcn_s_wait_tensorcnt(0);
#endif
    __syncthreads();

    #pragma unroll
    for (int ks = 0; ks < 64; ks += 32) {
      v16bf af[4], bfv[2];
      #pragma unroll
      for (int mi = 0; mi < 4; ++mi) af[mi] = load_frag_a(As + ks, wm + mi * 16, 64, lane);
      #pragma unroll
      for (int ni = 0; ni < 2; ++ni) bfv[ni] = load_frag_b(Bs + ks, wn + ni * 16, 64, lane);
      #pragma unroll
      for (int mi = 0; mi < 4; ++mi)
        #pragma unroll
        for (int ni = 0; ni < 2; ++ni)
          acc[mi][ni] = wmma_bf16(af[mi], bfv[ni], acc[mi][ni]);
    }
    __syncthreads();
  }

  // Epilogue: C element (v, lane): M = v + 8*(lane>=16), N = lane&15
  const int lh = lane >> 4, lm = lane & 15;
  #pragma unroll
  for (int mi = 0; mi < 4; ++mi)
    #pragma unroll
    for (int ni = 0; ni < 2; ++ni)
      #pragma unroll
      for (int v = 0; v < 8; ++v) {
        int row = bm + wm + mi * 16 + v + 8 * lh;
        int col = bn + wn + ni * 16 + lm;
        C[(size_t)row * N + col] = (OutT)acc[mi][ni][v];
      }
}

// ---------------------------------------------------------------------------
// RoPE in-place on q,k (bf16). One thread per (b*s, h, i<64) pair (i, i+64).
// ---------------------------------------------------------------------------
__global__ void __launch_bounds__(256) rope_kernel(bf16* __restrict__ q, bf16* __restrict__ k) {
  int t = blockIdx.x * 256 + threadIdx.x;
  int i  = t & 63;
  int h  = (t >> 6) & (H_ - 1);
  int bs = t >> 10;
  int s  = bs & (S_ - 1);
  float freq = __expf(-((float)(2 * i) * (1.0f / (float)HD_)) * 9.2103403720f); // ln(1e4)
  float ang = (float)s * freq;
  float sn, cs;
  __sincosf(ang, &sn, &cs);
  size_t base = (size_t)bs * D_ + h * HD_ + i;
  {
    float x1 = (float)q[base], x2 = (float)q[base + 64];
    q[base]      = (bf16)(x1 * cs - x2 * sn);
    q[base + 64] = (bf16)(x2 * cs + x1 * sn);
  }
  {
    float x1 = (float)k[base], x2 = (float)k[base + 64];
    k[base]      = (bf16)(x1 * cs - x2 * sn);
    k[base + 64] = (bf16)(x2 * cs + x1 * sn);
  }
}

// ---------------------------------------------------------------------------
// Flash attention (causal). Grid (S/128, H, B); 256 threads = 8 waves.
// Wave owns 16 query rows; K block (row-major) staged by TDM, V transposed
// manually; online softmax with ds_bpermute lane reductions.
// ---------------------------------------------------------------------------
__global__ void __launch_bounds__(256) attn_kernel(const bf16* __restrict__ q,
                                                   const bf16* __restrict__ k,
                                                   const bf16* __restrict__ v,
                                                   bf16* __restrict__ ctx) {
  __shared__ bf16 Ks[32 * 128];    // [key][d]
  __shared__ bf16 Vst[128 * 32];   // [d][key]
  __shared__ bf16 Ps[8][16 * 32];  // per-wave P relayout scratch

  const int tid  = threadIdx.x;
  const int lane = tid & 31;
  const int wave = tid >> 5;
  const int qblk = blockIdx.x * 128;
  const int h    = blockIdx.y;
  const int b    = blockIdx.z;
  const int row0 = qblk + wave * 16;
  const int lh = lane >> 4, lm = lane & 15;
  const float scale = 0.088388347648f;   // 1/sqrt(128)
#if HAVE_TDM
  const unsigned ldsK = lds_addr_of(&Ks[0]);
#endif

  v16bf qf[4];
  const size_t qbase = ((size_t)(b * S_ + row0)) * D_ + h * HD_;
  #pragma unroll
  for (int kc = 0; kc < 4; ++kc)
    qf[kc] = load_frag_a(q + qbase + kc * 32, 0, D_, lane);

  v8f o[8] = {};
  float m[8], l[8];
  #pragma unroll
  for (int i = 0; i < 8; ++i) { m[i] = -1e30f; l[i] = 0.0f; }

  const int jend = qblk + 127;
  for (int j0 = 0; j0 <= jend; j0 += 32) {
    // ---- Stage K block [j0..j0+31] x 128 via TDM ----
#if HAVE_TDM
    if (wave == 0)
      tdm_load_2d(ldsK, &k[((size_t)(b * S_ + j0)) * D_ + h * HD_],
                  /*tile_w=*/128, /*tile_h=*/32,
                  (unsigned long long)D_, (unsigned)D_, (unsigned)(1 << 20));
#else
    #pragma unroll
    for (int it = 0; it < 2; ++it) {
      int idx = (it * 256 + tid) * 8;
      int kr = idx >> 7, c = idx & 127;
      *(u32x4*)&Ks[idx] =
          *(const u32x4*)&k[((size_t)(b * S_ + j0 + kr)) * D_ + h * HD_ + c];
    }
#endif
    // ---- Stage V block transposed: Vst[d][key] ----
    #pragma unroll
    for (int it = 0; it < 2; ++it) {
      int idx = (it * 256 + tid) * 8;
      int vr = idx >> 7, c = idx & 127;
      u32x4 dv = *(const u32x4*)&v[((size_t)(b * S_ + j0 + vr)) * D_ + h * HD_ + c];
      const bf16* dp = (const bf16*)&dv;
      #pragma unroll
      for (int e = 0; e < 8; ++e) Vst[(c + e) * 32 + vr] = dp[e];
    }
#if HAVE_TDM
    if (wave == 0) __builtin_amdgcn_s_wait_tensorcnt(0);
#endif
    __syncthreads();

    if (j0 <= row0 + 15) {
      v8f s0 = {}, s1 = {};
      #pragma unroll
      for (int kc = 0; kc < 4; ++kc) {
        v16bf kf0 = load_frag_b(Ks + kc * 32, 0, 128, lane);
        v16bf kf1 = load_frag_b(Ks + kc * 32, 16, 128, lane);
        s0 = wmma_bf16(qf[kc], kf0, s0);
        s1 = wmma_bf16(qf[kc], kf1, s1);
      }
      #pragma unroll
      for (int vv = 0; vv < 8; ++vv) {
        int qi  = row0 + vv + 8 * lh;
        int kj0 = j0 + lm, kj1 = j0 + 16 + lm;
        float a0 = s0[vv] * scale;
        float a1 = s1[vv] * scale;
        if (kj0 > qi) a0 = -1e9f;
        if (kj1 > qi) a1 = -1e9f;
        float mx = fmaxf(a0, a1);
        #pragma unroll
        for (int ofs = 1; ofs < 16; ofs <<= 1) mx = fmaxf(mx, __shfl_xor(mx, ofs, 32));
        float newm = fmaxf(m[vv], mx);
        float corr = __expf(m[vv] - newm);
        float p0 = __expf(a0 - newm);
        float p1 = __expf(a1 - newm);
        float ps = p0 + p1;
        #pragma unroll
        for (int ofs = 1; ofs < 16; ofs <<= 1) ps += __shfl_xor(ps, ofs, 32);
        l[vv] = l[vv] * corr + ps;
        m[vv] = newm;
        #pragma unroll
        for (int nt = 0; nt < 8; ++nt) o[nt][vv] *= corr;
        Ps[wave][(vv + 8 * lh) * 32 + lm]      = (bf16)p0;
        Ps[wave][(vv + 8 * lh) * 32 + 16 + lm] = (bf16)p1;
      }
      v16bf pf = load_frag_a(&Ps[wave][0], 0, 32, lane);
      #pragma unroll
      for (int nt = 0; nt < 8; ++nt) {
        v16bf vf = load_frag_b(Vst, nt * 16, 32, lane);
        o[nt] = wmma_bf16(pf, vf, o[nt]);
      }
    }
    __syncthreads();
  }

  #pragma unroll
  for (int nt = 0; nt < 8; ++nt)
    #pragma unroll
    for (int vv = 0; vv < 8; ++vv) {
      int row = row0 + vv + 8 * lh;
      float ov = o[nt][vv] / l[vv];
      ctx[((size_t)(b * S_ + row)) * D_ + h * HD_ + nt * 16 + lm] = (bf16)ov;
    }
}

// ---------------------------------------------------------------------------
// Launch: cvt -> 3x GEMM -> rope -> attn -> GEMM
// Workspace (bf16 elems): xb | wqb wkb wvb wob | qb kb vb | ctxb (~117.5 MB)
// ---------------------------------------------------------------------------
extern "C" void kernel_launch(void* const* d_in, const int* in_sizes, int n_in,
                              void* d_out, int out_size, void* d_ws, size_t ws_size,
                              hipStream_t stream) {
  const float* x  = (const float*)d_in[0];
  const float* wq = (const float*)d_in[1];
  const float* wk = (const float*)d_in[2];
  const float* wv = (const float*)d_in[3];
  const float* wo = (const float*)d_in[4];
  float* out = (float*)d_out;

  const size_t NX = (size_t)B_ * S_ * D_;
  const size_t NW = (size_t)D_ * D_;

  bf16* ws  = (bf16*)d_ws;
  bf16* xb  = ws;
  bf16* wqb = xb  + NX;
  bf16* wkb = wqb + NW;
  bf16* wvb = wkb + NW;
  bf16* wob = wvb + NW;
  bf16* qb  = wob + NW;
  bf16* kb  = qb  + NX;
  bf16* vb  = kb  + NX;
  bf16* ctxb= vb  + NX;

  cvt_f32_bf16<<<(int)(NX / 4 + 255) / 256, 256, 0, stream>>>(x,  xb,  (int)(NX / 4));
  cvt_f32_bf16<<<(int)(NW / 4 + 255) / 256, 256, 0, stream>>>(wq, wqb, (int)(NW / 4));
  cvt_f32_bf16<<<(int)(NW / 4 + 255) / 256, 256, 0, stream>>>(wk, wkb, (int)(NW / 4));
  cvt_f32_bf16<<<(int)(NW / 4 + 255) / 256, 256, 0, stream>>>(wv, wvb, (int)(NW / 4));
  cvt_f32_bf16<<<(int)(NW / 4 + 255) / 256, 256, 0, stream>>>(wo, wob, (int)(NW / 4));

  dim3 gg(B_ * S_ / 128, D_ / 128);
  gemm_bf16_wmma<bf16><<<gg, 256, 0, stream>>>(xb, wqb, qb, B_ * S_, D_, D_);
  gemm_bf16_wmma<bf16><<<gg, 256, 0, stream>>>(xb, wkb, kb, B_ * S_, D_, D_);
  gemm_bf16_wmma<bf16><<<gg, 256, 0, stream>>>(xb, wvb, vb, B_ * S_, D_, D_);

  rope_kernel<<<(B_ * S_ * H_ * 64) / 256, 256, 0, stream>>>(qb, kb);

  dim3 ga(S_ / 128, H_, B_);
  attn_kernel<<<ga, 256, 0, stream>>>(qb, kb, vb, ctxb);

  gemm_bf16_wmma<float><<<gg, 256, 0, stream>>>(ctxb, wob, out, B_ * S_, D_, D_);
}
